// fused_classSR_3class_fsrcnn_net_48704929136844
// MI455X (gfx1250) — matrix operand order repaired and checked by reading, not compiled, not run on hardware
//
#include <hip/hip_runtime.h>

// ---------------------------------------------------------------------------
// ClassSR (classifier + 3x FSRCNN branches) for MI455X / gfx1250.
// Every conv is an implicit GEMM on v_wmma_f32_16x16x32_f16 (wave32).
// GEMM operands staged in LDS, zero-padded (K and spatial halo) so fragment
// gathers are unguarded DS loads; im2col offsets come from a precomputed LDS
// K-offset table hoisted into registers and reused across M-tiles.
// ---------------------------------------------------------------------------

typedef __attribute__((ext_vector_type(16))) _Float16 v16h;
typedef __attribute__((ext_vector_type(8)))  float    v8f;

#define NPATCH        512
#define OUT_PER_PATCH (3 * 128 * 128)

// ---- WMMA fragment lane layouts (wave32, per ISA) ----
// A 16x32: lane L: M = L&15;  half h -> K = h + 8*((L>>4) + (h>>3))
// B 32x16: lane L: N = L&15;  half h -> K = h + 8*((L>>4) + (h>>3))
// C 16x16: lane L: N = L&15;  vgpr v -> M = v + 8*(L>>4)
#define KIDX(h, g) ((h) + 8 * ((g) + ((h) >> 3)))

template <typename F>
__device__ inline v16h frag_a(F f) {
    const int lane = threadIdx.x & 31;
    const int g = lane >> 4, r = lane & 15;
    v16h a;
#pragma unroll
    for (int h = 0; h < 16; ++h)
        a[h] = f(r, KIDX(h, g));
    return a;
}

template <typename F>
__device__ inline v16h frag_b(F f) {
    const int lane = threadIdx.x & 31;
    const int g = lane >> 4, r = lane & 15;
    v16h b;
#pragma unroll
    for (int h = 0; h < 16; ++h)
        b[h] = f(KIDX(h, g), r);
    return b;
}

template <typename F>
__device__ inline void c_store(const v8f& acc, F f) {
    const int lane = threadIdx.x & 31;
    const int g = lane >> 4, r = lane & 15;
#pragma unroll
    for (int v = 0; v < 8; ++v)
        f(v + 8 * g, r, acc[v]);
}

__device__ inline v8f wmma_f16(const v16h& a, const v16h& b, const v8f& c) {
    return __builtin_amdgcn_wmma_f32_16x16x32_f16(
        false, a, false, b, (short)0, c, false, false);
}

__device__ inline float leaky01(float x) { return fmaxf(x, 0.1f * x); }
__device__ inline float prelu(float x, float a) {
    return fmaxf(x, 0.f) + a * fminf(x, 0.f);
}

// ===========================================================================
// Classifier. One block (8 wave32) per patch. M = 64 pixels.
// ===========================================================================
struct ClsParams {
    const float *c0_w, *c0_b, *c1_w, *c1_b, *c2_w, *c2_b, *c3_w, *c3_b,
                *c4_w, *c4_b, *fc_w, *fc_b;
};

__global__ __launch_bounds__(256) void classifier_kernel(
    const float* __restrict__ x, ClsParams P, int* __restrict__ assign) {
    __shared__ _Float16 act[2][64 * 128];   // 2 x 16KB ping-pong activations
    __shared__ _Float16 wbuf[8192];         // 16KB staged/padded weights
    __shared__ float feat[32];
    const int patch = blockIdx.x;
    const int tid = threadIdx.x;
    const int wv = tid >> 5;

    _Float16* xs = act[1];                  // 3072 halves (<= 8192 capacity)
    const float* xg = x + patch * 3072;
    for (int i = tid; i < 3072; i += 256) xs[i] = (_Float16)xg[i];
    for (int i = tid; i < 8192; i += 256) { // c0 W padded [128][64]
        int n = i >> 6, kk = i & 63;
        wbuf[i] = (kk < 48) ? (_Float16)P.c0_w[n * 48 + kk] : (_Float16)0.f;
    }
    __syncthreads();

    // ---- c0: 3->128 k4 s4.  M=64, N=128 (ntile = wave), K=48 pad 64 ----
    {
        const int nb = wv * 16;
        v8f acc[4] = {};
#pragma unroll
        for (int ks = 0; ks < 2; ++ks) {
            v16h bf = frag_b([&](int k, int n) -> _Float16 {
                return wbuf[(nb + n) * 64 + ks * 32 + k];
            });
#pragma unroll
            for (int mt = 0; mt < 4; ++mt) {
                v16h af = frag_a([&](int m, int k) -> _Float16 {
                    // k>=48 hits a zero B column; index stays < 8192.
                    int kk = ks * 32 + k;
                    int p = mt * 16 + m;
                    int oy = p >> 3, ox = p & 7;
                    int c = kk >> 4, t = kk & 15, ky = t >> 2, kx = t & 3;
                    return xs[c * 1024 + (oy * 4 + ky) * 32 + (ox * 4 + kx)];
                });
                acc[mt] = wmma_f16(af, bf, acc[mt]);
            }
        }
        __syncthreads();
#pragma unroll
        for (int mt = 0; mt < 4; ++mt)
            c_store(acc[mt], [&](int mm, int nn, float v) {
                int n = nb + nn;
                act[0][(mt * 16 + mm) * 128 + n] =
                    (_Float16)leaky01(v + P.c0_b[n]);
            });
    }
    __syncthreads();

    // ---- c1..c3: 1x1 128->128, LeakyReLU, K staged 32 wide per slab ----
    const float* ws[3] = {P.c1_w, P.c2_w, P.c3_w};
    const float* bs[3] = {P.c1_b, P.c2_b, P.c3_b};
    for (int L = 0; L < 3; ++L) {
        const _Float16* src = act[L & 1];
        _Float16* dst = act[(L + 1) & 1];
        const float* W = ws[L];
        const int nb = wv * 16;
        v8f acc[4] = {};
        for (int ks = 0; ks < 4; ++ks) {
            for (int i = tid; i < 4096; i += 256) {
                int n = i >> 5, k = i & 31;
                wbuf[i] = (_Float16)W[n * 128 + ks * 32 + k];
            }
            __syncthreads();
            v16h bf = frag_b([&](int k, int n) -> _Float16 {
                return wbuf[(nb + n) * 32 + k];
            });
#pragma unroll
            for (int mt = 0; mt < 4; ++mt) {
                v16h af = frag_a([&](int m, int k) -> _Float16 {
                    return src[(mt * 16 + m) * 128 + ks * 32 + k];
                });
                acc[mt] = wmma_f16(af, bf, acc[mt]);
            }
            __syncthreads();
        }
#pragma unroll
        for (int mt = 0; mt < 4; ++mt)
            c_store(acc[mt], [&](int mm, int nn, float v) {
                int n = nb + nn;
                dst[(mt * 16 + mm) * 128 + n] = (_Float16)leaky01(v + bs[L][n]);
            });
        __syncthreads();
    }

    // ---- c4: 1x1 128->32 (no act) ----
    float* c4buf = (float*)act[0];
    for (int i = tid; i < 4096; i += 256) wbuf[i] = (_Float16)P.c4_w[i];
    __syncthreads();
    {
        const int mt = wv >> 1, nb = (wv & 1) * 16;
        v8f acc = {};
#pragma unroll
        for (int ks = 0; ks < 4; ++ks) {
            v16h bf = frag_b([&](int k, int n) -> _Float16 {
                return wbuf[(nb + n) * 128 + ks * 32 + k];
            });
            v16h af = frag_a([&](int m, int k) -> _Float16 {
                return act[1][(mt * 16 + m) * 128 + ks * 32 + k];
            });
            acc = wmma_f16(af, bf, acc);
        }
        __syncthreads();
        c_store(acc, [&](int mm, int nn, float v) {
            c4buf[(mt * 16 + mm) * 32 + (nb + nn)] = v + P.c4_b[nb + nn];
        });
    }
    __syncthreads();

    if (tid < 32) {
        float s = 0.f;
        for (int m = 0; m < 64; ++m) s += c4buf[m * 32 + tid];
        feat[tid] = s * (1.f / 64.f);
    }
    __syncthreads();
    if (tid == 0) {
        float best = -1e30f; int bi = 0;
        for (int j = 0; j < 3; ++j) {
            float lg = P.fc_b[j];
            for (int n = 0; n < 32; ++n) lg += P.fc_w[j * 32 + n] * feat[n];
            if (lg > best) { best = lg; bi = j; }
        }
        assign[patch] = bi;
    }
}

// ===========================================================================
// FSRCNN branch. b6 (12->nf) folded into the deconv.
// Activations: zero-halo layout [34][34][12] f16; input [3][36][36] f16.
// ===========================================================================
struct BranchParams {
    const float *head_w, *head_b, *head_a;
    const float *b0_w, *b0_b;
    const float *bw1, *bb1, *bw2, *bb2, *bw3, *bb3, *bw4, *bb4;
    const float *body_a, *b6_w, *b6_b, *tail_w, *tail_b;
    int nf;
};

#define HBUF_BYTES  27776                    // (34*34*12 + 16 pad) halves
#define OFF_HBREG   27776                    // hB region base
#define SIZE_AB     55552                    // hA + hB
// head-phase overlay inside hB region:
#define OFF_HEADW   (OFF_HBREG + 0)          // [64][96] f16 = 12288B
#define OFF_XS      (OFF_HBREG + 12288)      // [3][36][36] f16 = 7776B
#define OFF_H1      (OFF_HBREG + 20064)      // [32][64] f16 = 4096B
#define OFF_B0W     (OFF_HBREG + 24160)      // [16][64] f16 = 2048B
// tail overlay inside (dead) hB region:
#define OFF_WEFF    OFF_HBREG                // f32[243*12] = 11664B
#define OFF_BEFF    (OFF_HBREG + 11664)      // f32[243]    =   972B
// extra region:
#define OFF_BODYW   55552                    // [16][128] f16 = 4096B
#define OFF_KOFFB   59648                    // int[128] = 512B
#define OFF_KOFFH   60160                    // int[96]  = 384B
#define OFF_HBIAS   60544                    // f32[64]
#define OFF_HPRELU  60800                    // f32[64]
#define OFF_B0BIAS  61056                    // f32[16]
#define OFF_BBIAS   61120                    // f32[16]
#define OFF_BPRELU  61184                    // f32[16]
#define OFF_DUMP    61248                    // 128B dead-store scratch
#define SMEM_BYTES  61376

__global__ __launch_bounds__(256) void fsrcnn_kernel(
    const float* __restrict__ x, const int* __restrict__ assign, int branch,
    BranchParams P, float* __restrict__ out) {
    const int patch = blockIdx.x;
    if (assign[patch] != branch) return;     // uniform scalar branch

    __shared__ __align__(16) unsigned char smem[SMEM_BYTES];
    _Float16* hA = (_Float16*)smem;                       // [34*34][12]
    _Float16* hB = (_Float16*)(smem + OFF_HBREG);
    _Float16* headWp = (_Float16*)(smem + OFF_HEADW);
    _Float16* xs = (_Float16*)(smem + OFF_XS);
    _Float16* H1 = (_Float16*)(smem + OFF_H1);
    _Float16* b0Wp = (_Float16*)(smem + OFF_B0W);
    _Float16* dump = (_Float16*)(smem + OFF_DUMP);
    int* koff_b = (int*)(smem + OFF_KOFFB);
    int* koff_h = (int*)(smem + OFF_KOFFH);
    float* hbias  = (float*)(smem + OFF_HBIAS);
    float* hprelu = (float*)(smem + OFF_HPRELU);
    float* b0bias = (float*)(smem + OFF_B0BIAS);
    float* bbias  = (float*)(smem + OFF_BBIAS);
    float* bprelu = (float*)(smem + OFF_BPRELU);

    const int tid = threadIdx.x, wv = tid >> 5;
    const int lane = tid & 31, g = lane >> 4, r = lane & 15;
    const int nf = P.nf;
    const int ntilesH = (nf + 15) >> 4;

    const float* xg = x + patch * 3072;
    __builtin_prefetch(xg, 0, 3);                         // global_prefetch_b8

    // ---- zero hA + hB (halos must be zero), int4-wide -----------------------
    {
        int4 z; z.x = z.y = z.z = z.w = 0;
        int4* z4 = (int4*)smem;
        for (int i = tid; i < SIZE_AB / 16; i += 256) z4[i] = z;
    }
    __syncthreads();

    // ---- stage head operands, K-offset tables, biases -----------------------
    for (int i = tid; i < 64 * 96; i += 256) {            // head W [64][96]
        int n = i / 96, kk = i - n * 96;
        headWp[i] = (n < nf && kk < 75) ? (_Float16)P.head_w[n * 75 + kk]
                                        : (_Float16)0.f;
    }
    for (int i = tid; i < 16 * 64; i += 256) {            // b0 W [16][64]
        int n = i >> 6, kk = i & 63;
        b0Wp[i] = (n < 12 && kk < nf) ? (_Float16)P.b0_w[n * nf + kk]
                                      : (_Float16)0.f;
    }
    for (int i = tid; i < 3072; i += 256) {               // xs interior
        int c = i >> 10, rem = i & 1023, iy = rem >> 5, ix = rem & 31;
        xs[c * 1296 + (iy + 2) * 36 + (ix + 2)] = (_Float16)xg[i];
    }
    if (tid < 96) {                                       // head im2col offsets
        int kk = tid, o = 0;
        if (kk < 75) {
            int c = kk / 25, t = kk - c * 25, ky = t / 5, kx = t - ky * 5;
            o = c * 1296 + ky * 36 + kx;
        }
        koff_h[tid] = o;
    }
    if (tid < 128) {                                      // body im2col offsets
        int kk = tid, o = 0;
        if (kk < 108) {
            int ch = kk / 9, t = kk - ch * 9;
            o = ((t / 3 - 1) * 34 + (t % 3 - 1)) * 12 + ch;
        }
        koff_b[tid] = o;
    }
    if (tid < 64) {
        hbias[tid]  = tid < nf ? P.head_b[tid] : 0.f;
        hprelu[tid] = tid < nf ? P.head_a[tid] : 0.f;
    }
    if (tid < 16) b0bias[tid] = tid < 12 ? P.b0_b[tid] : 0.f;
    __syncthreads();

    // ---- head (5x5, 3->nf, PReLU) fused with b0 (1x1, nf->12) ---------------
    int koh[48];
#pragma unroll
    for (int ks = 0; ks < 3; ++ks)
#pragma unroll
        for (int h = 0; h < 16; ++h)
            koh[ks * 16 + h] = koff_h[ks * 32 + KIDX(h, g)];

    for (int mb = 0; mb < 32; ++mb) {                     // 32 pixels per iter
        const int T = 2 * ntilesH;
        if (wv < T) {
            const int mt = wv & 1, nb = (wv >> 1) * 16;
            const int p = mb * 32 + mt * 16 + r;
            const int base = (p >> 5) * 36 + (p & 31);
            v8f acc = {};
#pragma unroll
            for (int ks = 0; ks < 3; ++ks) {              // K = 75 -> 96
                v16h bf = frag_b([&](int k, int n) -> _Float16 {
                    return headWp[(nb + n) * 96 + ks * 32 + k];
                });
                v16h af;
#pragma unroll
                for (int h = 0; h < 16; ++h) af[h] = xs[base + koh[ks * 16 + h]];
                acc = wmma_f16(af, bf, acc);
            }
            c_store(acc, [&](int mm, int nn, float v) {
                int n = nb + nn;
                float val = prelu(v + hbias[n], hprelu[n]);
                _Float16* pp = (n < nf) ? (H1 + (mt * 16 + mm) * 64 + n)
                                        : (dump + (tid & 63));
                *pp = (_Float16)val;
            });
        }
        __syncthreads();

        if (wv < 2) {                                     // b0: nf->12
            const int mt = wv;
            const int ksteps = (nf > 32) ? 2 : 1;
            v8f acc = {};
            for (int ks = 0; ks < ksteps; ++ks) {
                v16h bf = frag_b([&](int k, int n) -> _Float16 {
                    return b0Wp[n * 64 + ks * 32 + k];
                });
                v16h af = frag_a([&](int m, int k) -> _Float16 {
                    return H1[(mt * 16 + m) * 64 + ks * 32 + k];
                });
                acc = wmma_f16(af, bf, acc);
            }
            c_store(acc, [&](int mm, int nn, float v) {
                int pix = mb * 32 + mt * 16 + mm;
                int pb = ((pix >> 5) + 1) * 34 + ((pix & 31) + 1);
                _Float16* pp = (nn < 12) ? (hA + pb * 12 + nn)
                                         : (dump + (tid & 63));
                *pp = (_Float16)(v + b0bias[nn]);
            });
        }
        __syncthreads();
    }

    // ---- re-zero hB (head overlay trashed its halo) ------------------------
    {
        int4 z; z.x = z.y = z.z = z.w = 0;
        int4* z4 = (int4*)(smem + OFF_HBREG);
        for (int i = tid; i < HBUF_BYTES / 16; i += 256) z4[i] = z;
    }
    __syncthreads();

    // ---- body: 4x conv3x3 12->12, PReLU folded into layer 4 -----------------
    _Float16* bodyWp = (_Float16*)(smem + OFF_BODYW);
    const float* bwL[4] = {P.bw1, P.bw2, P.bw3, P.bw4};
    const float* bbL[4] = {P.bb1, P.bb2, P.bb3, P.bb4};
    for (int L = 0; L < 4; ++L) {
        const _Float16* src = (L & 1) ? hB : hA;
        _Float16* dst = (L & 1) ? hA : hB;                // final lands in hA
        for (int i = tid; i < 16 * 128; i += 256) {       // stage [16][128]
            int n = i >> 7, kk = i & 127;
            bodyWp[i] = (n < 12 && kk < 108) ? (_Float16)bwL[L][n * 108 + kk]
                                             : (_Float16)0.f;
        }
        if (tid < 16) {
            bbias[tid]  = tid < 12 ? bbL[L][tid] : 0.f;
            bprelu[tid] = tid < 12 ? P.body_a[tid] : 0.f;
        }
        __syncthreads();

        v8f accs[8] = {};
#pragma unroll
        for (int ks = 0; ks < 4; ++ks) {
            v16h bf = frag_b([&](int k, int n) -> _Float16 {
                return bodyWp[n * 128 + ks * 32 + k];
            });
            int ko[16];
#pragma unroll
            for (int h = 0; h < 16; ++h)
                ko[h] = koff_b[ks * 32 + KIDX(h, g)];     // hoisted, reused 8x
#pragma unroll
            for (int j = 0; j < 8; ++j) {
                const int p = (wv + j * 8) * 16 + r;
                const int base = (((p >> 5) + 1) * 34 + ((p & 31) + 1)) * 12;
                v16h af;
#pragma unroll
                for (int h = 0; h < 16; ++h) af[h] = src[base + ko[h]];
                accs[j] = wmma_f16(af, bf, accs[j]);
            }
        }
        const bool doPrelu = (L == 3);
#pragma unroll
        for (int j = 0; j < 8; ++j) {
            const int mt = wv + j * 8;
            c_store(accs[j], [&](int mm, int nn, float v) {
                float val = v + bbias[nn];
                if (doPrelu) val = prelu(val, bprelu[nn]);
                int pix = mt * 16 + mm;
                int pb = ((pix >> 5) + 1) * 34 + ((pix & 31) + 1);
                _Float16* pp = (nn < 12) ? (dst + pb * 12 + nn)
                                         : (dump + (tid & 63));
                *pp = (_Float16)val;
            });
        }
        __syncthreads();
    }

    // ---- fold b6 into deconv weights (into dead hB region) ------------------
    float* Weff = (float*)(smem + OFF_WEFF);              // [243][12]
    float* beff = (float*)(smem + OFF_BEFF);              // [243]
    for (int e = tid; e < 243; e += 256) {                // e = c*81 + tap
        float wm[12];
#pragma unroll
        for (int m = 0; m < 12; ++m) wm[m] = 0.f;
        float bacc = 0.f;
        for (int n = 0; n < nf; ++n) {
            float wvn = P.tail_w[n * 243 + e];
            bacc += wvn * P.b6_b[n];
#pragma unroll
            for (int m = 0; m < 12; ++m) wm[m] += wvn * P.b6_w[n * 12 + m];
        }
#pragma unroll
        for (int m = 0; m < 12; ++m) Weff[e * 12 + m] = wm[m];
        beff[e] = bacc;
    }
    __syncthreads();

    // ---- deconv: k9 s4 p4 outpad3 -> 3x128x128, float4 stores ---------------
    float4* po4 = (float4*)(out + (size_t)patch * OUT_PER_PATCH);
    const float tb0 = P.tail_b[0], tb1 = P.tail_b[1], tb2 = P.tail_b[2];
    for (int chunk = tid; chunk < OUT_PER_PATCH / 4; chunk += 256) {
        int c = chunk >> 12;
        int rem = chunk & 4095;
        int oy = rem >> 5;
        int ox0 = (rem & 31) * 4;
        int iy_lo = (oy <= 4) ? 0 : ((oy - 1) >> 2);      // ceil((oy-4)/4)
        int iy_hi = (oy + 4) >> 2; if (iy_hi > 31) iy_hi = 31;
        float res[4];
        float tb = c == 0 ? tb0 : (c == 1 ? tb1 : tb2);
#pragma unroll
        for (int e = 0; e < 4; ++e) {
            int ox = ox0 + e;
            int ix_lo = (ox <= 4) ? 0 : ((ox - 1) >> 2);
            int ix_hi = (ox + 4) >> 2; if (ix_hi > 31) ix_hi = 31;
            float acc = tb;
            for (int iy = iy_lo; iy <= iy_hi; ++iy) {
                int ky = oy + 4 - 4 * iy;
                for (int ix = ix_lo; ix <= ix_hi; ++ix) {
                    int kx = ox + 4 - 4 * ix;
                    int tap = c * 81 + ky * 9 + kx;
                    float a = beff[tap];
                    const float* wp = Weff + tap * 12;
                    const _Float16* hp = hA + ((iy + 1) * 34 + (ix + 1)) * 12;
#pragma unroll
                    for (int m = 0; m < 12; ++m) a += wp[m] * (float)hp[m];
                    acc += a;
                }
            }
            res[e] = acc;
        }
        float4 v; v.x = res[0]; v.y = res[1]; v.z = res[2]; v.w = res[3];
        po4[chunk] = v;
    }
}

// ===========================================================================
__global__ __launch_bounds__(256) void counts_kernel(
    const int* __restrict__ assign, float* __restrict__ oc) {
    __shared__ int c[3];
    if (threadIdx.x < 3) c[threadIdx.x] = 0;
    __syncthreads();
    for (int i = threadIdx.x; i < NPATCH; i += 256) atomicAdd(&c[assign[i]], 1);
    __syncthreads();
    if (threadIdx.x < 3) oc[threadIdx.x] = (float)c[threadIdx.x];
}

// ===========================================================================
// Input order: setup_inputs() insertion order:
//   0: x
//   cls: 1 c0_w, 2 c0_b, 3 c4_w, 4 c4_b, 5 fc_w, 6 fc_b,
//        7 c1_w, 8 c1_b, 9 c2_w, 10 c2_b, 11 c3_w, 12 c3_b
//   net{b} @ 13 + 18*b: head_w, head_b, head_a, b0_w, b0_b, body_a,
//        b6_w, b6_b, tail_w, tail_b, b1_w..b4_b
// ===========================================================================
extern "C" void kernel_launch(void* const* d_in, const int* in_sizes, int n_in,
                              void* d_out, int out_size, void* d_ws, size_t ws_size,
                              hipStream_t stream) {
    (void)in_sizes; (void)n_in; (void)out_size; (void)ws_size;

    const float* x = (const float*)d_in[0];

    ClsParams C;
    C.c0_w = (const float*)d_in[1];  C.c0_b = (const float*)d_in[2];
    C.c4_w = (const float*)d_in[3];  C.c4_b = (const float*)d_in[4];
    C.fc_w = (const float*)d_in[5];  C.fc_b = (const float*)d_in[6];
    C.c1_w = (const float*)d_in[7];  C.c1_b = (const float*)d_in[8];
    C.c2_w = (const float*)d_in[9];  C.c2_b = (const float*)d_in[10];
    C.c3_w = (const float*)d_in[11]; C.c3_b = (const float*)d_in[12];

    static const int NFS[3] = {16, 36, 56};
    BranchParams BP[3];
    for (int b = 0; b < 3; ++b) {
        void* const* q = d_in + 13 + b * 18;
        BranchParams& P = BP[b];
        P.head_w = (const float*)q[0];  P.head_b = (const float*)q[1];
        P.head_a = (const float*)q[2];
        P.b0_w = (const float*)q[3];    P.b0_b = (const float*)q[4];
        P.body_a = (const float*)q[5];
        P.b6_w = (const float*)q[6];    P.b6_b = (const float*)q[7];
        P.tail_w = (const float*)q[8];  P.tail_b = (const float*)q[9];
        P.bw1 = (const float*)q[10];    P.bb1 = (const float*)q[11];
        P.bw2 = (const float*)q[12];    P.bb2 = (const float*)q[13];
        P.bw3 = (const float*)q[14];    P.bb3 = (const float*)q[15];
        P.bw4 = (const float*)q[16];    P.bb4 = (const float*)q[17];
        P.nf = NFS[b];
    }

    int* assign = (int*)d_ws;
    float* out = (float*)d_out;

    classifier_kernel<<<NPATCH, 256, 0, stream>>>(x, C, assign);
    for (int b = 0; b < 3; ++b)
        fsrcnn_kernel<<<NPATCH, 256, 0, stream>>>(x, assign, b, BP[b], out);
    counts_kernel<<<1, 256, 0, stream>>>(assign, out + (size_t)NPATCH * OUT_PER_PATCH);
}
